// MixerDiffAttention_63127429317263
// MI455X (gfx1250) — matrix operand order repaired and checked by reading, not compiled, not run on hardware
//
#include <hip/hip_runtime.h>
#include <hip/hip_bf16.h>

// ---------------------------------------------------------------------------
// MixerDiffAttention for MI455X (gfx1250, wave32, WMMA)
// ---------------------------------------------------------------------------
#define B_  2
#define T_  2048
#define D_  2048
#define NH  16
#define HD  256
#define H2  8
#define VD  512          // 2*HD, v head dim
#define NDIM (NH * HD)   // 4096

typedef __attribute__((ext_vector_type(16))) __bf16 v16bf;
typedef __attribute__((ext_vector_type(8)))  __bf16 v8bf;
typedef __attribute__((ext_vector_type(2)))  __bf16 v2bf;
typedef __attribute__((ext_vector_type(8)))  float  v8f;

static_assert(sizeof(__bf16) == 2, "bf16 size");

#if defined(__has_builtin)
#if __has_builtin(__builtin_amdgcn_cvt_pk_bf16_f32)
#define HAVE_PK_BF16 1
#endif
#if __has_builtin(__builtin_amdgcn_perm)
#define HAVE_PERM 1
#endif
#endif

__device__ __forceinline__ __bf16 f2bf(float f) {
  unsigned u = __builtin_bit_cast(unsigned, f);
  u += 0x7FFFu + ((u >> 16) & 1u);               // round-to-nearest-even
  unsigned short h = (unsigned short)(u >> 16);
  return __builtin_bit_cast(__bf16, h);
}

// Packed 2x fp32 -> bf16: prefer v_cvt_pk_bf16_f32, else one v_perm_b32 after
// a +0x8000 round bias (round-half-away; fine here), else scalar RNE.
__device__ __forceinline__ v2bf f2bf2(float lo, float hi) {
#if defined(HAVE_PK_BF16)
  return __builtin_amdgcn_cvt_pk_bf16_f32(lo, hi);
#elif defined(HAVE_PERM)
  const unsigned a = __builtin_bit_cast(unsigned, lo) + 0x8000u;
  const unsigned b = __builtin_bit_cast(unsigned, hi) + 0x8000u;
  // dst = { hi16(b), hi16(a) } : byte sel 7,6 from S0(=b), 3,2 from S1(=a)
  const unsigned p = __builtin_amdgcn_perm(b, a, 0x07060302u);
  return __builtin_bit_cast(v2bf, p);
#else
  v2bf r; r[0] = f2bf(lo); r[1] = f2bf(hi); return r;
#endif
}

__device__ __forceinline__ v8f zero8() {
  v8f z;
#pragma unroll
  for (int i = 0; i < 8; ++i) z[i] = 0.0f;
  return z;
}

__device__ __forceinline__ v8f wmma_bf16(v16bf a, v16bf b, v8f c) {
  // (neg_a, A, neg_b, B, c_mod, C, reuse_a, reuse_b)
  return __builtin_amdgcn_wmma_f32_16x16x32_bf16(false, a, false, b,
                                                 (short)0, c, false, false);
}

// A-matrix fragment, 16x32 bf16, row-major LDS tile with leading dim `ld`.
// ISA layout: lanes 0-15 -> M=lane, K runs [0,8)+[16,24); lanes 16-31 -> K runs [8,16)+[24,32).
__device__ __forceinline__ v16bf ldsA16x32(const __bf16* tile, int ld, int lane) {
  const int r  = lane & 15;
  const int kb = (lane >> 4) << 3;          // 0 or 8
  v16bf a;
  ((v8bf*)&a)[0] = *(const v8bf*)(tile + r * ld + kb);
  ((v8bf*)&a)[1] = *(const v8bf*)(tile + r * ld + kb + 16);
  return a;
}

// B-matrix fragment, 32x16 bf16. Source stored "N-major": value(k,n) = colTile[n*ld + k].
// ISA layout: lanes 0-15 -> N=lane, K runs [0,8)+[16,24); lanes 16-31 -> K runs [8,16)+[24,32).
__device__ __forceinline__ v16bf ldsB32x16(const __bf16* colTile, int ld, int lane) {
  const int n  = lane & 15;
  const int kb = (lane >> 4) << 3;          // 0 or 8
  v16bf b;
  ((v8bf*)&b)[0] = *(const v8bf*)(colTile + n * ld + kb);
  ((v8bf*)&b)[1] = *(const v8bf*)(colTile + n * ld + kb + 16);
  return b;
}

// ---------------------------------------------------------------------------
// One-shot fp32 -> bf16 conversion (x, Wq, Wk, Wv, v). 8 elems/thread,
// packed v_perm conversion, single b128 store. grid.x = n/2048.
// ---------------------------------------------------------------------------
__global__ __launch_bounds__(256)
void cvt_f32_bf16(const float* __restrict__ in, __bf16* __restrict__ out)
{
  const size_t base = ((size_t)blockIdx.x * 256 + threadIdx.x) * 8;
  v8bf o;
#pragma unroll
  for (int u = 0; u < 8; u += 2)
    ((v2bf*)&o)[u >> 1] = f2bf2(in[base + u], in[base + u + 1]);
  *(v8bf*)(out + base) = o;
}

// ---------------------------------------------------------------------------
// GEMM: C[M=4096, N=4096] = A[4096, 2048] x W[4096, 2048]^T
// bf16 in (pre-converted), fp32 out. block = 256 threads (8 waves),
// tile 64x128, K-step 64. LDS fills are pure b128 copies; inner loop is
// ds_load_b128 pairs + 8 WMMA per wave per barrier round.
// ---------------------------------------------------------------------------
__global__ __launch_bounds__(256)
void gemm_bf16(const __bf16* __restrict__ A, const __bf16* __restrict__ W,
               float* __restrict__ C)
{
  __shared__ __align__(16) __bf16 sA[64 * 64];     //  8 KB
  __shared__ __align__(16) __bf16 sW[128 * 64];    // 16 KB

  const int tid  = threadIdx.x;
  const int lane = tid & 31;
  const int wid  = tid >> 5;
  const int wm   = wid >> 1;                 // 0..3  (16-row slice)
  const int wn   = wid & 1;                  // 0..1  (64-col slice)
  const int mBase = blockIdx.y * 64;
  const int nBase = blockIdx.x * 128;

  v8f acc[4];
#pragma unroll
  for (int i = 0; i < 4; ++i) acc[i] = zero8();

  for (int k0 = 0; k0 < D_; k0 += 64) {
    __syncthreads();
    {   // A tile fill: 64x64 bf16, 2 x b128 per thread
      const int r  = tid >> 2;
      const int c0 = (tid & 3) * 16;
      const __bf16* src = A + (size_t)(mBase + r) * D_ + k0 + c0;
      if (k0 + 64 < D_) __builtin_prefetch(src + 64, 0, 1);
#pragma unroll
      for (int u = 0; u < 2; ++u)
        *(v8bf*)(sA + r * 64 + c0 + u * 8) = *(const v8bf*)(src + u * 8);
    }
    {   // W tile fill: 128x64 bf16, 4 x b128 per thread
      const int r  = tid >> 1;
      const int c0 = (tid & 1) * 32;
      const __bf16* src = W + (size_t)(nBase + r) * D_ + k0 + c0;
      if (k0 + 64 < D_) __builtin_prefetch(src + 64, 0, 1);
#pragma unroll
      for (int u = 0; u < 4; ++u)
        *(v8bf*)(sW + r * 64 + c0 + u * 8) = *(const v8bf*)(src + u * 8);
    }
    __syncthreads();

#pragma unroll
    for (int kk = 0; kk < 2; ++kk) {
      const v16bf a = ldsA16x32(sA + (wm * 16) * 64 + kk * 32, 64, lane);
#pragma unroll
      for (int nt = 0; nt < 4; ++nt) {
        const v16bf bw = ldsB32x16(sW + (wn * 64 + nt * 16) * 64 + kk * 32, 64, lane);
        acc[nt] = wmma_bf16(a, bw, acc[nt]);
      }
    }
  }

  // C-layout store: row = v + 8*(lane>=16), col = lane&15 within each 16-tile
  const int lan = lane & 15;
  const int hs  = lane >> 4;
#pragma unroll
  for (int nt = 0; nt < 4; ++nt) {
#pragma unroll
    for (int v = 0; v < 8; ++v) {
      const int row = mBase + wm * 16 + v + (hs << 3);
      const int col = nBase + wn * 64 + nt * 16 + lan;
      C[(size_t)row * NDIM + col] = acc[nt][v];
    }
  }
}

// ---------------------------------------------------------------------------
// Fused RMSNorm + RoPE over one 256-wide head row; emits bf16.
// grid.x = B*T*NH, block = 128 (thread i handles pair (i, i+128)).
// ---------------------------------------------------------------------------
__global__ __launch_bounds__(128)
void rmsnorm_rope(const float* __restrict__ X, __bf16* __restrict__ O)
{
  const int row = blockIdx.x;                  // ((b*T + t)*NH + head)
  const int t   = (row >> 4) & (T_ - 1);
  const int i   = threadIdx.x;

  const float* xr = X + (size_t)row * HD;
  const float x1 = xr[i];
  const float x2 = xr[i + 128];

  float ss = x1 * x1 + x2 * x2;
#pragma unroll
  for (int off = 16; off >= 1; off >>= 1) ss += __shfl_xor(ss, off, 32);
  __shared__ float red[4];
  if ((i & 31) == 0) red[i >> 5] = ss;
  __syncthreads();
  const float tot  = red[0] + red[1] + red[2] + red[3];
  const float rinv = rsqrtf(tot * (1.0f / 256.0f) + 1.1920929e-07f);

  const float invf = __powf(10000.0f, -(float)i * (1.0f / 128.0f));
  float sn, cs;
  __sincosf((float)t * invf, &sn, &cs);

  const float a = x1 * rinv;
  const float b = x2 * rinv;
  O[(size_t)row * HD + i]       = f2bf(a * cs + b * sn);
  O[(size_t)row * HD + i + 128] = f2bf(-a * sn + b * cs);
}

// ---------------------------------------------------------------------------
// lambda = exp(sum q1*k1) - exp(sum q2*k2) + 0.2
// ---------------------------------------------------------------------------
__global__ __launch_bounds__(128)
void lambda_kernel(const float* __restrict__ q1, const float* __restrict__ k1,
                   const float* __restrict__ q2, const float* __restrict__ k2,
                   float* __restrict__ lam)
{
  const int i = threadIdx.x;                   // 0..127 == HD/2
  float a = q1[i] * k1[i];
  float b = q2[i] * k2[i];
#pragma unroll
  for (int off = 16; off >= 1; off >>= 1) {
    a += __shfl_xor(a, off, 32);
    b += __shfl_xor(b, off, 32);
  }
  __shared__ float ra[4], rb[4];
  if ((i & 31) == 0) { ra[i >> 5] = a; rb[i >> 5] = b; }
  __syncthreads();
  if (i == 0) {
    const float sa = ra[0] + ra[1] + ra[2] + ra[3];
    const float sb = rb[0] + rb[1] + rb[2] + rb[3];
    *lam = __expf(sa) - __expf(sb) + 0.2f;     // LAMBDA_INIT = 0.2
  }
}

// ---------------------------------------------------------------------------
// Causal flash attention, one branch. Block = 128 threads (4 waves).
// Block tile: 16 query rows x full 512 v-cols of one (b, vhead); each wave owns
// 128 v-cols. Online softmax in fp32; QK^T and PV via bf16 WMMA
// (24 WMMA per wave per 32-key block).
// ---------------------------------------------------------------------------
__global__ __launch_bounds__(128)
void attn_kernel(const __bf16* __restrict__ Q, const __bf16* __restrict__ K,
                 const __bf16* __restrict__ V, float* __restrict__ Y, int branch)
{
  __shared__ __align__(16) __bf16 sQ[16 * 256];     //  8 KB
  __shared__ __align__(16) __bf16 sK[32 * 256];     // 16 KB
  __shared__ __align__(16) __bf16 sVT[512 * 32];    // 32 KB, transposed: [n*32 + k]
  __shared__ __align__(16) __bf16 sP[4][16 * 32];   //  4 KB, per-wave P scratch

  const int tid  = threadIdx.x;
  const int lane = tid & 31;
  const int wid  = tid >> 5;
  const int lan  = lane & 15;
  const int hs   = lane >> 4;                       // half select (C-layout row +8)

  const int iBase = blockIdx.x * 16;
  const int vh    = blockIdx.y;
  const int b     = blockIdx.z;
  const int head  = vh + branch * H2;

  {   // Q tile: 16x256, copy bf16
    const int r  = tid >> 3;
    const int c0 = (tid & 7) * 32;
    const __bf16* src = Q + ((size_t)(b * T_ + iBase + r) * NH + head) * HD + c0;
#pragma unroll
    for (int u = 0; u < 4; ++u)
      *(v8bf*)(sQ + r * 256 + c0 + u * 8) = *(const v8bf*)(src + u * 8);
  }

  v8f acc[8];
#pragma unroll
  for (int i = 0; i < 8; ++i) acc[i] = zero8();
  float mrow[8], lrow[8];
#pragma unroll
  for (int i = 0; i < 8; ++i) { mrow[i] = -1.0e30f; lrow[i] = 0.0f; }

  const int nkb = (iBase + 15) / 32 + 1;            // causal key-block count

  for (int jb = 0; jb < nkb; ++jb) {
    const int jBase = jb * 32;
    __syncthreads();
    {   // K tile: 32x256, row-major copy
      const int r  = tid >> 2;
      const int c0 = (tid & 3) * 64;
      const __bf16* src = K + ((size_t)(b * T_ + jBase + r) * NH + head) * HD + c0;
#pragma unroll
      for (int u = 0; u < 8; ++u)
        *(v8bf*)(sK + r * 256 + c0 + u * 8) = *(const v8bf*)(src + u * 8);
    }
    {   // V tile: 32x512, transpose into sVT[n*32 + k]
      const int r  = tid >> 2;
      const int c0 = (tid & 3) * 128;
      const __bf16* src = V + ((size_t)(b * T_ + jBase + r) * H2 + vh) * VD + c0;
#pragma unroll
      for (int u = 0; u < 16; ++u) {
        const v8bf vv = *(const v8bf*)(src + u * 8);
#pragma unroll
        for (int e = 0; e < 8; ++e)
          sVT[(c0 + u * 8 + e) * 32 + r] = vv[e];
      }
    }
    __syncthreads();

    // S = Q * K^T : two 16x16 C tiles (key halves), K=256 in 8 WMMA steps each
    v8f s0 = zero8(), s1 = zero8();
#pragma unroll
    for (int kk = 0; kk < 8; ++kk) {
      const v16bf a  = ldsA16x32(sQ + kk * 32, 256, lane);
      const v16bf b0 = ldsB32x16(sK + kk * 32, 256, lane);            // keys 0..15
      const v16bf b1 = ldsB32x16(sK + 16 * 256 + kk * 32, 256, lane); // keys 16..31
      s0 = wmma_bf16(a, b0, s0);
      s1 = wmma_bf16(a, b1, s1);
    }

    // Online softmax (fp32), C-layout: row = v + 8*hs, col = lan (+16 for s1)
    __bf16* sPw = sP[wid];
#pragma unroll
    for (int v = 0; v < 8; ++v) {
      const int row = iBase + v + (hs << 3);
      float e0 = s0[v] * 0.0625f;                  // 1/sqrt(256)
      float e1 = s1[v] * 0.0625f;
      if (jBase + lan      > row) e0 = -3.0e38f;   // causal mask
      if (jBase + 16 + lan > row) e1 = -3.0e38f;

      float rm = fmaxf(e0, e1);
#pragma unroll
      for (int off = 8; off >= 1; off >>= 1) rm = fmaxf(rm, __shfl_xor(rm, off, 32));
      const float mnew = fmaxf(mrow[v], rm);
      const float corr = __expf(mrow[v] - mnew);
      const float p0 = __expf(e0 - mnew);
      const float p1 = __expf(e1 - mnew);
      float rs = p0 + p1;
#pragma unroll
      for (int off = 8; off >= 1; off >>= 1) rs += __shfl_xor(rs, off, 32);
      lrow[v] = lrow[v] * corr + rs;
      mrow[v] = mnew;
#pragma unroll
      for (int nt = 0; nt < 8; ++nt) acc[nt][v] *= corr;

      const int rl = v + (hs << 3);                // C-layout -> row-major P in LDS
      sPw[rl * 32 + lan]      = f2bf(p0);
      sPw[rl * 32 + 16 + lan] = f2bf(p1);
    }

    // acc += P(16x32) * V(32x128 per wave), 8 WMMA per wave
    const v16bf pA = ldsA16x32(sPw, 32, lane);
#pragma unroll
    for (int nt = 0; nt < 8; ++nt) {
      const v16bf bV = ldsB32x16(sVT + (wid * 128 + nt * 16) * 32, 32, lane);
      acc[nt] = wmma_bf16(pA, bV, acc[nt]);
    }
  }

  // Epilogue: divide by row sum, fp32 store
#pragma unroll
  for (int v = 0; v < 8; ++v) {
    const float inv = 1.0f / lrow[v];
    const int t = iBase + v + (hs << 3);
    float* dst = Y + ((size_t)(b * T_ + t) * H2 + vh) * VD + wid * 128 + lan;
#pragma unroll
    for (int nt = 0; nt < 8; ++nt)
      dst[nt * 16] = acc[nt][v] * inv;
  }
}

// ---------------------------------------------------------------------------
// out = y1 - lam * y2
// ---------------------------------------------------------------------------
__global__ __launch_bounds__(256)
void combine_kernel(const float* __restrict__ y1, const float* __restrict__ y2,
                    const float* __restrict__ lam, float* __restrict__ out)
{
  const float l = *lam;
  const size_t base = ((size_t)blockIdx.x * 256 + threadIdx.x) * 8;
#pragma unroll
  for (int u = 0; u < 8; ++u) out[base + u] = y1[base + u] - l * y2[base + u];
}

// ---------------------------------------------------------------------------
extern "C" void kernel_launch(void* const* d_in, const int* in_sizes, int n_in,
                              void* d_out, int out_size, void* d_ws, size_t ws_size,
                              hipStream_t stream)
{
  (void)in_sizes; (void)n_in; (void)out_size; (void)ws_size;

  const float* x   = (const float*)d_in[0];
  const float* Wq  = (const float*)d_in[1];
  const float* Wk  = (const float*)d_in[2];
  const float* Wv  = (const float*)d_in[3];
  const float* lq1 = (const float*)d_in[4];
  const float* lk1 = (const float*)d_in[5];
  const float* lq2 = (const float*)d_in[6];
  const float* lk2 = (const float*)d_in[7];
  float* out = (float*)d_out;

  const size_t NELEM = (size_t)B_ * T_ * NDIM;      // 16M elements (q/k/v/y)
  const size_t NELX  = (size_t)B_ * T_ * D_;        //  8M elements (x)
  const size_t NELW  = (size_t)NDIM * D_;           //  8M elements (each W)

  char* w = (char*)d_ws;
  float*  lam   = (float*)w;  w += 256;
  float*  q_raw = (float*)w;  w += NELEM * sizeof(float);
  float*  k_raw = (float*)w;  w += NELEM * sizeof(float);
  float*  v_raw = (float*)w;  w += NELEM * sizeof(float);
  __bf16* qb    = (__bf16*)w; w += NELEM * sizeof(__bf16);
  __bf16* kb    = (__bf16*)w; w += NELEM * sizeof(__bf16);
  __bf16* vb    = (__bf16*)w; w += NELEM * sizeof(__bf16);
  __bf16* xb    = (__bf16*)w; w += NELX  * sizeof(__bf16);
  __bf16* wqb   = (__bf16*)w; w += NELW  * sizeof(__bf16);
  __bf16* wkb   = (__bf16*)w; w += NELW  * sizeof(__bf16);
  __bf16* wvb   = (__bf16*)w; w += NELW  * sizeof(__bf16);
  // fp32 intermediates are dead after the bf16 versions exist -> reuse as y1/y2
  float* y1 = v_raw;
  float* y2 = q_raw;

  // One-shot fp32 -> bf16 of the GEMM operands (removes all conversion work
  // from the GEMM inner loops; x/W were otherwise re-converted per block).
  cvt_f32_bf16<<<(unsigned)(NELX / 2048), 256, 0, stream>>>(x,  xb);
  cvt_f32_bf16<<<(unsigned)(NELW / 2048), 256, 0, stream>>>(Wq, wqb);
  cvt_f32_bf16<<<(unsigned)(NELW / 2048), 256, 0, stream>>>(Wk, wkb);
  cvt_f32_bf16<<<(unsigned)(NELW / 2048), 256, 0, stream>>>(Wv, wvb);

  const dim3 gGemm(NDIM / 128, (B_ * T_) / 64);     // 32 x 64 blocks
  gemm_bf16<<<gGemm, 256, 0, stream>>>(xb, wqb, q_raw);
  gemm_bf16<<<gGemm, 256, 0, stream>>>(xb, wkb, k_raw);
  gemm_bf16<<<gGemm, 256, 0, stream>>>(xb, wvb, v_raw);

  rmsnorm_rope<<<B_ * T_ * NH, 128, 0, stream>>>(q_raw, qb);
  rmsnorm_rope<<<B_ * T_ * NH, 128, 0, stream>>>(k_raw, kb);
  cvt_f32_bf16<<<(unsigned)(NELEM / 2048), 256, 0, stream>>>(v_raw, vb);
  lambda_kernel<<<1, 128, 0, stream>>>(lq1, lk1, lq2, lk2, lam);

  const dim3 gAttn(T_ / 16, H2, B_);
  attn_kernel<<<gAttn, 128, 0, stream>>>(qb, kb, vb, y1, 0);
  attn_kernel<<<gAttn, 128, 0, stream>>>(qb, kb, vb, y2, 1);

  combine_kernel<<<(unsigned)(NELEM / 2048), 256, 0, stream>>>(y1, y2, lam, out);
}
